// PMA_69818988364321
// MI455X (gfx1250) — matrix operand order repaired and checked by reading, not compiled, not run on hardware
//
#include <hip/hip_runtime.h>
#include <hip/hip_bf16.h>
#include <stdint.h>

// ---------------------------------------------------------------------------
// PMA (pooled multi-head attention) for MI455X / gfx1250, wave32 + WMMA bf16.
//
// Algebraic refactoring (memory-roofline driven, HBM = 23.3 TB/s):
//   scores = X @ P^T            with P[hk,:] = q[kk,h,:] @ Wk_head   (16 x 256)
//            (bk-derived per-row constant dropped: softmax shift-invariant)
//   Z[b,hk,:] = sum_l attn[b,hk,l] * X[b,l,:]
//   out = Z @ Wv^T + bv  (since sum_l attn == 1), then o_proj + LayerNorm.
// K/V projections (68.7 GFLOP + >0.5 GB traffic) are never materialized;
// total traffic ~0.57 GB -> ~25 us HBM-bound.
// Dual WMMA accumulator chains break the D->D RAW hazard (no v_nop stalls).
// ---------------------------------------------------------------------------

typedef __bf16 v16bf __attribute__((ext_vector_type(16)));
typedef float  v8f   __attribute__((ext_vector_type(8)));

#define B_      64
#define L_      4096
#define D_      256
#define K_      4
#define H_      4
#define HK_     16
#define NSPLIT  8
#define NEGV    (-1e9f)
#define TSTRIDE 36   // LDS row stride in bf16 elems: conflict-free, 8B-aligned rows

union Frag16 {
  unsigned short s[16];
  __bf16         b[16];
  uint2          u2[4];
  uint4          u4[2];
  v16bf          v;
};

// ---------------------------------------------------------------------------
// Kernel 1: q = S@Wq^T + bq ; P[hk,:] = q[kk, h*64:(h+1)*64] @ Wk_head  (bf16)
// ---------------------------------------------------------------------------
__global__ void k_prep(const float* __restrict__ S, const float* __restrict__ Wq,
                       const float* __restrict__ bq, const float* __restrict__ Wk,
                       __bf16* __restrict__ Pbf) {
  __shared__ float qs[K_][D_];
  const int t = threadIdx.x;                 // 256 threads
  for (int kk = 0; kk < K_; ++kk) {
    float acc = bq[t];
    const float* wrow = Wq + (size_t)t * D_;
    const float* srow = S + (size_t)kk * D_;
    for (int i = 0; i < D_; ++i) acc += srow[i] * wrow[i];
    qs[kk][t] = acc;
  }
  __syncthreads();
  for (int hk = 0; hk < HK_; ++hk) {
    const int h = hk >> 2, kk = hk & 3;
    float acc = 0.f;
    for (int dd = 0; dd < 64; ++dd)
      acc += qs[kk][h * 64 + dd] * Wk[(size_t)(h * 64 + dd) * D_ + t];
    Pbf[hk * D_ + t] = (__bf16)acc;
  }
}

// ---------------------------------------------------------------------------
// Kernel 2: SC[b][l][hk] = X[b,l,:] . P[hk,:]   (16-row tile per wave, WMMA)
// ---------------------------------------------------------------------------
__global__ void k_scores(const float* __restrict__ X,
                         const __bf16* __restrict__ Pbf,
                         float* __restrict__ SC) {
  const int wave = threadIdx.x >> 5;
  const int lane = threadIdx.x & 31;
  const int half = lane >> 4;        // 0: lanes 0-15, 1: lanes 16-31
  const int l16  = lane & 15;
  const int b    = blockIdx.x >> 5;
  const int l0   = ((blockIdx.x & 31) << 7) + (wave << 4);

  // Preload loop-invariant B fragments (P^T). Lane n holds col hk=n:
  //   lanes 0-15: K = s*32 .. s*32+15 ; lanes 16-31: K = s*32+16 .. s*32+31
  v16bf bf[8];
  for (int s = 0; s < 8; ++s)
    bf[s] = *reinterpret_cast<const v16bf*>(Pbf + l16 * D_ + s * 32 + half * 16);

  // Two independent accumulator chains -> consecutive WMMAs have no RAW
  // hazard on D (avoids the 1+4-NOP penalty from ISA 7.12.1).
  v8f c0, c1;
  for (int i = 0; i < 8; ++i) { c0[i] = 0.f; c1[i] = 0.f; }

  const float* xrow = X + ((size_t)b * L_ + (size_t)(l0 + l16)) * D_;
#pragma unroll
  for (int s = 0; s < 8; ++s) {
    // A (16x32 bf16): lane m (0-15): K c0..c0+15, c0=s*32
    //                 lane m+16   : shifted by 16.
    const int cc = s * 32 + half * 16;
    float4 f0 = *reinterpret_cast<const float4*>(xrow + cc);
    float4 f1 = *reinterpret_cast<const float4*>(xrow + cc + 4);
    float4 f2 = *reinterpret_cast<const float4*>(xrow + cc + 8);
    float4 f3 = *reinterpret_cast<const float4*>(xrow + cc + 12);
    Frag16 a;
    a.b[0]  = (__bf16)f0.x; a.b[1]  = (__bf16)f0.y; a.b[2]  = (__bf16)f0.z; a.b[3]  = (__bf16)f0.w;
    a.b[4]  = (__bf16)f1.x; a.b[5]  = (__bf16)f1.y; a.b[6]  = (__bf16)f1.z; a.b[7]  = (__bf16)f1.w;
    a.b[8]  = (__bf16)f2.x; a.b[9]  = (__bf16)f2.y; a.b[10] = (__bf16)f2.z; a.b[11] = (__bf16)f2.w;
    a.b[12] = (__bf16)f3.x; a.b[13] = (__bf16)f3.y; a.b[14] = (__bf16)f3.z; a.b[15] = (__bf16)f3.w;
    if (s & 1)
      c1 = __builtin_amdgcn_wmma_f32_16x16x32_bf16(false, a.v, false, bf[s],
                                                   (short)0, c1, false, false);
    else
      c0 = __builtin_amdgcn_wmma_f32_16x16x32_bf16(false, a.v, false, bf[s],
                                                   (short)0, c0, false, false);
  }
  c0 = c0 + c1;
  // D (16x16 f32): VGPR r = row r (lanes 0-15) / row 8+r (lanes 16-31), N=lane
  for (int r = 0; r < 8; ++r) {
    const int row = l0 + half * 8 + r;
    SC[((size_t)b * L_ + row) * HK_ + l16] = c0[r];
  }
}

// ---------------------------------------------------------------------------
// Kernel 3: masked softmax over L per (b,hk); writes attn as bf16 [b][hk][l]
// ---------------------------------------------------------------------------
__global__ void k_softmax(const float* __restrict__ SC,
                          const unsigned char* __restrict__ mask,
                          __bf16* __restrict__ AT) {
  const int b = blockIdx.x;
  __shared__ int s_any;
  if (threadIdx.x == 0) s_any = 0;
  __syncthreads();
  int anyv = 0;
  for (int l = threadIdx.x; l < L_; l += blockDim.x)
    anyv |= (int)mask[(size_t)b * L_ + l];
  if (anyv) atomicOr(&s_any, 1);
  __syncthreads();
  const bool allpad = (s_any == 0);

  const int w = threadIdx.x >> 5;    // hk
  const int lane = threadIdx.x & 31;
  const float* scp = SC + (size_t)b * L_ * HK_ + w;
  const unsigned char* mp = mask + (size_t)b * L_;

  float mx = -3.4e38f;
  for (int l = lane; l < L_; l += 32) {
    bool pad = (mp[l] == 0);
    if (allpad && l == 0) pad = false;
    float v = pad ? NEGV : scp[(size_t)l * HK_] * 0.125f;  // 1/sqrt(dh=64)
    mx = fmaxf(mx, v);
  }
  for (int off = 16; off; off >>= 1) mx = fmaxf(mx, __shfl_xor(mx, off, 32));

  float sum = 0.f;
  for (int l = lane; l < L_; l += 32) {
    bool pad = (mp[l] == 0);
    if (allpad && l == 0) pad = false;
    float v = pad ? NEGV : scp[(size_t)l * HK_] * 0.125f;
    sum += __expf(v - mx);
  }
  for (int off = 16; off; off >>= 1) sum += __shfl_xor(sum, off, 32);
  const float inv = 1.f / sum;

  __bf16* atp = AT + ((size_t)b * HK_ + w) * L_;
  for (int l = lane; l < L_; l += 32) {
    bool pad = (mp[l] == 0);
    if (allpad && l == 0) pad = false;
    float v = pad ? NEGV : scp[(size_t)l * HK_] * 0.125f;
    atp[l] = (__bf16)(__expf(v - mx) * inv);
  }
}

// ---------------------------------------------------------------------------
// Kernel 4: ZP[b][split][hk][d] = attn(16 x Lsplit) @ X(Lsplit x 256) via WMMA
//           X tile staged transposed to LDS (bf16, padded stride).
// ---------------------------------------------------------------------------
__global__ void k_zpass(const float* __restrict__ X,
                        const __bf16* __restrict__ AT,
                        float* __restrict__ ZP) {
  __shared__ __bf16 tile[D_ * TSTRIDE];   // 18 KB
  const int b     = blockIdx.x >> 3;
  const int split = blockIdx.x & 7;
  const int wave  = threadIdx.x >> 5;  // 16 waves, wave -> 16-col slab
  const int lane  = threadIdx.x & 31;
  const int half  = lane >> 4;
  const int l16   = lane & 15;
  const int n0    = wave << 4;

  v8f c0, c1;
  for (int i = 0; i < 8; ++i) { c0[i] = 0.f; c1[i] = 0.f; }

  const int kbeg = split * (L_ / NSPLIT);
  for (int k0 = kbeg; k0 < kbeg + L_ / NSPLIT; k0 += 64) {
    // Two 32-wide k-steps per trip, alternating accumulators.
#pragma unroll
    for (int sub = 0; sub < 2; ++sub) {
      const int kk0 = k0 + sub * 32;
      // Stage X[b][kk0:kk0+32][0:256] -> LDS transposed: tile[d*36 + l_local]
      for (int e = threadIdx.x; e < 32 * D_; e += blockDim.x) {
        const int ll = e >> 8;
        const int d  = e & 255;
        tile[d * TSTRIDE + ll] = (__bf16)X[((size_t)b * L_ + (kk0 + ll)) * D_ + d];
      }
      __syncthreads();

      // A (attn 16x32 bf16), rows = hk: lane m: K kk0+half*16 .. kk0+half*16+15
      Frag16 a;
      const __bf16* atp = AT + ((size_t)b * HK_ + l16) * L_;
      a.u4[0] = *reinterpret_cast<const uint4*>(atp + kk0 + half * 16);
      a.u4[1] = *reinterpret_cast<const uint4*>(atp + kk0 + 8 + half * 16);

      // B (X 32x16 bf16): lane n: col d=n0+n, K 0..15 (lanes 0-15) / 16..31
      Frag16 bb;
      const uint2* bp =
          reinterpret_cast<const uint2*>(&tile[(n0 + l16) * TSTRIDE + half * 16]);
      bb.u2[0] = bp[0]; bb.u2[1] = bp[1]; bb.u2[2] = bp[2]; bb.u2[3] = bp[3];

      if (sub)
        c1 = __builtin_amdgcn_wmma_f32_16x16x32_bf16(false, a.v, false, bb.v,
                                                     (short)0, c1, false, false);
      else
        c0 = __builtin_amdgcn_wmma_f32_16x16x32_bf16(false, a.v, false, bb.v,
                                                     (short)0, c0, false, false);
      __syncthreads();
    }
  }
  c0 = c0 + c1;
  for (int r = 0; r < 8; ++r) {
    const int hk = half * 8 + r;
    ZP[(((size_t)b * NSPLIT + split) * HK_ + hk) * D_ + n0 + l16] = c0[r];
  }
}

// ---------------------------------------------------------------------------
// Kernel 5: Z-partial reduce, out = Z@Wv^T+bv, y = out@Wo^T+bo, LayerNorm
// ---------------------------------------------------------------------------
__global__ void k_final(const float* __restrict__ ZP, const float* __restrict__ Wv,
                        const float* __restrict__ bv, const float* __restrict__ Wo,
                        const float* __restrict__ bo, const float* __restrict__ gamma,
                        const float* __restrict__ beta, float* __restrict__ out) {
  const int b  = blockIdx.x >> 2;
  const int kk = blockIdx.x & 3;
  const int t  = threadIdx.x;          // 256 threads
  __shared__ float Zs[H_][D_];
  __shared__ float outs[D_];
  __shared__ float ys[D_];
  __shared__ float red[256];

  for (int h = 0; h < H_; ++h) {
    const int hk = h * K_ + kk;
    float acc = 0.f;
    for (int s = 0; s < NSPLIT; ++s)
      acc += ZP[(((size_t)b * NSPLIT + s) * HK_ + hk) * D_ + t];
    Zs[h][t] = acc;
  }
  __syncthreads();
  {
    const int h = t >> 6;              // head of output dim t
    float acc = bv[t];
    const float* wrow = Wv + (size_t)t * D_;
    for (int i = 0; i < D_; ++i) acc += Zs[h][i] * wrow[i];
    outs[t] = acc;
  }
  __syncthreads();
  {
    float acc = bo[t];
    const float* wrow = Wo + (size_t)t * D_;
    for (int i = 0; i < D_; ++i) acc += outs[i] * wrow[i];
    ys[t] = acc;
  }
  __syncthreads();
  red[t] = ys[t];
  __syncthreads();
  for (int off = 128; off; off >>= 1) {
    if (t < off) red[t] += red[t + off];
    __syncthreads();
  }
  const float mu = red[0] * (1.0f / D_);
  const float d0 = ys[t] - mu;
  __syncthreads();
  red[t] = d0 * d0;
  __syncthreads();
  for (int off = 128; off; off >>= 1) {
    if (t < off) red[t] += red[t + off];
    __syncthreads();
  }
  const float var = red[0] * (1.0f / D_);
  const float rs  = rsqrtf(var + 1e-5f);
  out[((size_t)b * K_ + kk) * D_ + t] = d0 * rs * gamma[t] + beta[t];
}

// ---------------------------------------------------------------------------
extern "C" void kernel_launch(void* const* d_in, const int* in_sizes, int n_in,
                              void* d_out, int out_size, void* d_ws, size_t ws_size,
                              hipStream_t stream) {
  (void)in_sizes; (void)n_in; (void)out_size; (void)ws_size;
  const float*         X    = (const float*)d_in[0];
  const unsigned char* mask = (const unsigned char*)d_in[1];  // jnp bool = 1 byte
  const float*         S    = (const float*)d_in[2];
  const float*         Wq   = (const float*)d_in[3];
  const float*         bq   = (const float*)d_in[4];
  const float*         Wk   = (const float*)d_in[5];
  // d_in[6] = bk: provably unused (softmax shift invariance)
  const float*         Wv   = (const float*)d_in[7];
  const float*         bv   = (const float*)d_in[8];
  const float*         Wo   = (const float*)d_in[9];
  const float*         bo   = (const float*)d_in[10];
  const float*         gam  = (const float*)d_in[11];
  const float*         bet  = (const float*)d_in[12];
  float*               out  = (float*)d_out;

  char* ws = (char*)d_ws;
  __bf16* Pbf = (__bf16*)ws;                                              // 8 KB
  float*  SC  = (float*)(ws + 8192);                                      // 16 MB
  __bf16* AT  = (__bf16*)(ws + 8192 + (size_t)B_ * L_ * HK_ * 4);         // 8 MB
  float*  ZP  = (float*)((char*)AT + (size_t)B_ * HK_ * L_ * 2);          // 8 MB

  k_prep   <<<1,                256, 0, stream>>>(S, Wq, bq, Wk, Pbf);
  k_scores <<<B_ * (L_ / 128),  256, 0, stream>>>(X, Pbf, SC);
  k_softmax<<<B_,               512, 0, stream>>>(SC, mask, AT);
  k_zpass  <<<B_ * NSPLIT,      512, 0, stream>>>(X, AT, ZP);
  k_final  <<<B_ * K_,          256, 0, stream>>>(ZP, Wv, bv, Wo, bo, gam, bet, out);
}
